// Cache_14413910245413
// MI455X (gfx1250) — matrix operand order repaired and checked by reading, not compiled, not run on hardware
//
#include <hip/hip_runtime.h>
#include <hip/hip_bf16.h>
#include <math.h>

typedef __attribute__((ext_vector_type(16))) __bf16 v16bf;
typedef __attribute__((ext_vector_type(8)))  __bf16 v8bf;
typedef __attribute__((ext_vector_type(4)))  __bf16 v4bf;
typedef __attribute__((ext_vector_type(8)))  float  v8f;

#define L     64
#define NHID  256
#define NBLK  512
#define BSZ   32
#define TOPK  16

// LDS row stride for keys tile (bf16 elems): 256 + 8 pad to break bank conflicts
#define KSTRIDE 264

// Native conversion: lets clang emit hardware bf16 converts (packed cvt on gfx1250)
__device__ __forceinline__ __bf16 f2bf(float f) { return (__bf16)f; }

__device__ __forceinline__ v4bf cvt4(float4 v) {
    v4bf o;
    o.x = (__bf16)v.x; o.y = (__bf16)v.y; o.z = (__bf16)v.z; o.w = (__bf16)v.w;
    return o;
}

// One workgroup per (b, n): compute max_{i,j} q[b,i,:]·k[n,b,j,:]
// 128 threads = 4 waves; wave w handles M rows [16w, 16w+16).
__global__ __launch_bounds__(128) void cache_attn_max_kernel(
    const float* __restrict__ query,   // (1, L, BSZ, NHID)
    const float* __restrict__ keys,    // (NBLK, BSZ, L*NHID)
    float* __restrict__ attention)     // (BSZ, 1, NBLK)
{
    __shared__ __bf16 kb[L * KSTRIDE];   // keys tile as bf16, padded rows
    __shared__ float  wgmax[4];

    const int bn   = blockIdx.x;
    const int b    = bn >> 9;            // / NBLK
    const int n    = bn & (NBLK - 1);
    const int tid  = threadIdx.x;
    const int wave = tid >> 5;
    const int lane = tid & 31;

    // ---- cooperative load of keys tile (64x256 f32, contiguous) -> bf16 LDS ----
    const float* ktile = keys + ((size_t)n * BSZ + b) * (size_t)(L * NHID);
    const float4* k4 = (const float4*)ktile;
    #pragma unroll
    for (int it = 0; it < 32; ++it) {
        int f4 = tid + it * 128;            // 4096 float4 total
        float4 v = k4[f4];
        int felem = f4 << 2;                // float index
        int row = felem >> 8;               // / NHID
        int col = felem & (NHID - 1);
        *(v4bf*)&kb[row * KSTRIDE + col] = cvt4(v);  // 8B aligned (KSTRIDE%4==0)
    }

    // ---- preload A fragments (this wave's 16x256 strip of q) ----
    // A layout (16-bit 16x32): lane = {half=lane>>4, m=lane&15};
    // elems 0..15 -> K = 16*half + t
    const int m    = (wave << 4) + (lane & 15);
    const int half = lane >> 4;
    const float* qrow = query + ((size_t)m * BSZ + b) * NHID;
    v16bf afrag[8];
    #pragma unroll
    for (int kk = 0; kk < 8; ++kk) {
        int h0 = (kk << 5) + (half << 4);
        const float4* qa = (const float4*)(qrow + h0);
        const float4* qb = (const float4*)(qrow + h0 + 8);
        v4bf c0 = cvt4(qa[0]), c1 = cvt4(qa[1]);
        v4bf c2 = cvt4(qb[0]), c3 = cvt4(qb[1]);
        v16bf a;
        ((v4bf*)&a)[0] = c0;
        ((v4bf*)&a)[1] = c1;
        ((v4bf*)&a)[2] = c2;
        ((v4bf*)&a)[3] = c3;
        afrag[kk] = a;
    }

    __syncthreads();

    // ---- GEMM strips + running max ----
    // B layout (16-bit 32x16): lane = {khalf=lane>>4, ncol=lane&15};
    // elems 0..15 -> K = 16*khalf + t  at column ncol.
    const int ncol  = lane & 15;
    const int khalf = lane >> 4;
    float smax = -INFINITY;

    #pragma unroll
    for (int jt = 0; jt < 4; ++jt) {
        const int j0 = jt << 4;
        v8f c = {};
        #pragma unroll
        for (int kk = 0; kk < 8; ++kk) {
            const int col = (kk << 5) + (khalf << 4);
            const __bf16* src = &kb[(j0 + ncol) * KSTRIDE + col]; // 16B aligned
            v16bf bfr;
            ((v8bf*)&bfr)[0] = ((const v8bf*)src)[0];
            ((v8bf*)&bfr)[1] = ((const v8bf*)src)[1];
            c = __builtin_amdgcn_wmma_f32_16x16x32_bf16(
                    false, afrag[kk], false, bfr, (short)0, c, false, false);
        }
        #pragma unroll
        for (int r = 0; r < 8; ++r) smax = fmaxf(smax, c[r]);
    }

    // wave32 max reduction
    #pragma unroll
    for (int off = 16; off > 0; off >>= 1)
        smax = fmaxf(smax, __shfl_xor(smax, off, 32));

    if (lane == 0) wgmax[wave] = smax;
    __syncthreads();
    if (tid == 0) {
        float mm = fmaxf(fmaxf(wgmax[0], wgmax[1]), fmaxf(wgmax[2], wgmax[3]));
        attention[b * NBLK + n] = mm;
    }
}

// One workgroup per batch: top-16 (descending, ties -> lowest index) over 512 values.
__global__ __launch_bounds__(256) void cache_topk_kernel(
    const float* __restrict__ att,     // (BSZ, 1, NBLK)
    float* __restrict__ out_idx)       // (TOPK, BSZ, 1), indices stored as float
{
    __shared__ float sv[NBLK];
    __shared__ float rv[256];
    __shared__ int   ri[256];

    const int b = blockIdx.x;
    const int t = threadIdx.x;

    sv[t]       = att[b * NBLK + t];
    sv[t + 256] = att[b * NBLK + t + 256];
    __syncthreads();

    for (int kk = 0; kk < TOPK; ++kk) {
        float v0 = sv[t], v1 = sv[t + 256];
        // prefer higher value; on tie prefer lower index (i0 < i1 always here)
        bool take1 = (v1 > v0);
        rv[t] = take1 ? v1 : v0;
        ri[t] = take1 ? (t + 256) : t;
        __syncthreads();
        #pragma unroll
        for (int s = 128; s > 0; s >>= 1) {
            if (t < s) {
                float a = rv[t], c = rv[t + s];
                int ia = ri[t], ic = ri[t + s];
                if (c > a || (c == a && ic < ia)) { rv[t] = c; ri[t] = ic; }
            }
            __syncthreads();
        }
        if (t == 0) {
            out_idx[kk * BSZ + b] = (float)ri[0];
            sv[ri[0]] = -INFINITY;   // remove winner
        }
        __syncthreads();
    }
}

extern "C" void kernel_launch(void* const* d_in, const int* in_sizes, int n_in,
                              void* d_out, int out_size, void* d_ws, size_t ws_size,
                              hipStream_t stream) {
    const float* query = (const float*)d_in[0];   // (1, 64, 32, 256)
    const float* keys  = (const float*)d_in[1];   // (512, 32, 16384)
    float* out = (float*)d_out;                   // [16384 attention | 512 indices]

    cache_attn_max_kernel<<<dim3(BSZ * NBLK), dim3(128), 0, stream>>>(query, keys, out);
    cache_topk_kernel<<<dim3(BSZ), dim3(256), 0, stream>>>(out, out + BSZ * NBLK);
}
